// HardMoe_10153302688475
// MI455X (gfx1250) — matrix operand order repaired and checked by reading, not compiled, not run on hardware
//
#include <hip/hip_runtime.h>
#include <hip/hip_bf16.h>

#define NTOK   16384   // N*S tokens
#define DIM    2048    // D
#define OUTD   2048    // O
#define NEXP   8       // E experts
#define MAXTOK 16384   // max tokens per expert list

typedef __attribute__((ext_vector_type(16))) _Float16 v16h;
typedef __attribute__((ext_vector_type(8)))  _Float16 v8h;
typedef __attribute__((ext_vector_type(4)))  _Float16 v4h;
typedef __attribute__((ext_vector_type(8)))  float    v8f;
typedef __attribute__((ext_vector_type(4)))  float    v4f;

// ---------------- workspace layout ----------------
// [0,32)            : per-expert counters (8 x int)
// [256, 256+512KB)  : per-expert token lists (8 x 16384 int)
// [1MB, 1MB+64MB)   : x converted to f16, row-major [token][D]
// [1MB+64MB, +64MB) : expert_w packed into WMMA B fragments (f16)
#define WS_COUNTS 0
#define WS_LISTS  256
#define WS_XF16   ((size_t)1 << 20)
#define WS_WPACK  (WS_XF16 + (size_t)NTOK * DIM * 2)

// ---------------- init: zero out + counters ----------------
__global__ void k_init(float* __restrict__ out, int* __restrict__ counts) {
    size_t i = (size_t)blockIdx.x * blockDim.x + threadIdx.x;
    size_t n4 = (size_t)NTOK * OUTD / 4;
    if (i < n4) ((v4f*)out)[i] = (v4f){0.f, 0.f, 0.f, 0.f};
    if (blockIdx.x == 0 && threadIdx.x < NEXP) counts[threadIdx.x] = 0;
}

// ---------------- convert x: f32 -> f16 ----------------
__global__ void k_cvt_x(const float* __restrict__ x, _Float16* __restrict__ xf) {
    size_t i = ((size_t)blockIdx.x * blockDim.x + threadIdx.x) * 4;
    if (i >= (size_t)NTOK * DIM) return;
    v4f v = *(const v4f*)(x + i);
    v4h o = {(_Float16)v.x, (_Float16)v.y, (_Float16)v.z, (_Float16)v.w};
    *(v4h*)(xf + i) = o;
}

// ---------------- pack expert_w into WMMA B-fragment layout ----------------
// B is 32(K) x 16(N) per fragment. Lane L holds column n=L%16,
// K = 16*(L/16) + t, t=0..15 (VGPR t/2, half t%2). Fragment = 512 f16,
// stored lane-major so the GEMM reads 32 contiguous bytes per lane.
// Source reads are coalesced; scattered 2B writes merge in L2 (fragment
// regions stay resident across the k-filling blocks).
__global__ void k_pack_w(const float* __restrict__ w, _Float16* __restrict__ wp) {
    size_t i = (size_t)blockIdx.x * blockDim.x + threadIdx.x;
    if (i >= (size_t)NEXP * DIM * OUTD) return;
    int n = (int)(i % OUTD);
    size_t r = i / OUTD;
    int k = (int)(r % DIM);
    int e = (int)(r / DIM);
    int kb = k >> 5, ko = k & 31;
    int half = ko >> 4, t = ko & 15;
    int lane = half * 16 + (n & 15);
    int ntile = n >> 4;                       // 0..127
    size_t frag = ((size_t)e * (OUTD / 16) + ntile) * (DIM / 32) + kb;
    wp[frag * 512 + lane * 16 + t] = (_Float16)w[i];
}

// ---------------- gating: logits + top-2 + route ----------------
// one wave (32 lanes) per token; coalesced over D; shfl_xor reduction
__global__ void k_gate(const float* __restrict__ x, const float* __restrict__ gw,
                       const float* __restrict__ gb, int* __restrict__ counts,
                       int* __restrict__ lists) {
    int wave = threadIdx.x >> 5, lane = threadIdx.x & 31;
    int tok = blockIdx.x * 8 + wave;
    const float* xr = x + (size_t)tok * DIM;
    float acc[NEXP];
#pragma unroll
    for (int e = 0; e < NEXP; e++) acc[e] = 0.f;
    for (int d = lane; d < DIM; d += 32) {
        float xv = xr[d];
        v4f g0 = *(const v4f*)(gw + (size_t)d * NEXP);
        v4f g1 = *(const v4f*)(gw + (size_t)d * NEXP + 4);
        acc[0] += xv * g0.x; acc[1] += xv * g0.y;
        acc[2] += xv * g0.z; acc[3] += xv * g0.w;
        acc[4] += xv * g1.x; acc[5] += xv * g1.y;
        acc[6] += xv * g1.z; acc[7] += xv * g1.w;
    }
#pragma unroll
    for (int off = 16; off >= 1; off >>= 1)
#pragma unroll
        for (int e = 0; e < NEXP; e++) acc[e] += __shfl_xor(acc[e], off, 32);
    if (lane == 0) {
        float best = -3.4e38f, sec = -3.4e38f;
        int bi = 0, si = 0;
#pragma unroll
        for (int e = 0; e < NEXP; e++) {
            float v = acc[e] + gb[e];
            if (v > best) { sec = best; si = bi; best = v; bi = e; }
            else if (v > sec) { sec = v; si = e; }
        }
        int p0 = atomicAdd(&counts[bi], 1); lists[bi * MAXTOK + p0] = tok;
        int p1 = atomicAdd(&counts[si], 1); lists[si * MAXTOK + p1] = tok;
    }
}

// ---------------- sparse expert GEMM: Y += 0.5*relu(Xg @ We + be) ----------------
// block: 256 thr = 8 waves (2x4), block tile 128(M) x 256(N);
// wave tile 64 x 64 -> 16 WMMA : 16 b128 loads per K-step (K-step = 32)
__global__ __launch_bounds__(256)
void k_moe_gemm(const _Float16* __restrict__ xf, const _Float16* __restrict__ wp,
                const float* __restrict__ eb, const int* __restrict__ counts,
                const int* __restrict__ lists, float* __restrict__ out) {
    int bx = blockIdx.x;
    int nb = bx & 7;            // 8 N-blocks of 256
    int mb = (bx >> 3) & 127;   // 128 M-blocks of 128
    int e  = bx >> 10;          // expert
    int ne = counts[e];
    int mbase = mb * 128;
    if (mbase >= ne) return;    // uniform per block: EXEC stays full for WMMA

    int tid = threadIdx.x;
    int w = tid >> 5, lane = tid & 31;
    int wm = w & 1, wn = w >> 1;       // 2 x 4 wave grid
    int h = lane >> 4, l16 = lane & 15;
    const int* list = lists + (size_t)e * MAXTOK;

    // A row base pointers (k order used here: lane holds K=16h..16h+15, matching packed B)
    // clamp row index -> unconditional, branch-free gather of token ids
    const _Float16* aBase[4];
#pragma unroll
    for (int ms = 0; ms < 4; ms++) {
        int row = mbase + wm * 64 + ms * 16 + l16;
        int rc = (row < ne) ? row : (ne - 1);
        int tok = list[rc];
        aBase[ms] = xf + (size_t)tok * DIM + h * 16;
    }
    // B fragment base pointer; ns-subtiles are 64 frags (64KB) apart -> imm offsets
    int ntile0 = nb * 16 + wn * 4;
    const _Float16* bBase =
        wp + (((size_t)e * (OUTD / 16) + ntile0) * (DIM / 32)) * 512 + lane * 16;

    v8f acc[4][4];
#pragma unroll
    for (int ms = 0; ms < 4; ms++)
#pragma unroll
        for (int ns = 0; ns < 4; ns++)
            acc[ms][ns] = (v8f){0.f, 0.f, 0.f, 0.f, 0.f, 0.f, 0.f, 0.f};

#pragma unroll 1
    for (int kb = 0; kb < DIM / 32; kb++) {
        v16h a[4], b[4];
#pragma unroll
        for (int ms = 0; ms < 4; ms++) {
            v8h a0 = *(const v8h*)(aBase[ms] + kb * 32);
            v8h a1 = *(const v8h*)(aBase[ms] + kb * 32 + 8);
            a[ms] = __builtin_shufflevector(a0, a1, 0, 1, 2, 3, 4, 5, 6, 7,
                                            8, 9, 10, 11, 12, 13, 14, 15);
        }
#pragma unroll
        for (int ns = 0; ns < 4; ns++) {
            const _Float16* bp = bBase + ((size_t)ns * (DIM / 32) + kb) * 512;
            v8h b0 = *(const v8h*)(bp);
            v8h b1 = *(const v8h*)(bp + 8);
            b[ns] = __builtin_shufflevector(b0, b1, 0, 1, 2, 3, 4, 5, 6, 7,
                                            8, 9, 10, 11, 12, 13, 14, 15);
        }
#pragma unroll
        for (int ms = 0; ms < 4; ms++)
#pragma unroll
            for (int ns = 0; ns < 4; ns++)
                acc[ms][ns] = __builtin_amdgcn_wmma_f32_16x16x32_f16(
                    false, a[ms], false, b[ns], (short)0, acc[ms][ns], false, false);
    }

    // epilogue: bias + relu + 0.5x, atomic scatter-add per token row
    // branch-free token gather (clamped index), predicate only on the atomic
#pragma unroll
    for (int ms = 0; ms < 4; ms++) {
        int rbase = mbase + wm * 64 + ms * 16 + 8 * h;  // C layout: VGPR r -> M = r + 8h
        int toks[8];
        bool vrow[8];
#pragma unroll
        for (int r = 0; r < 8; r++) {
            int row = rbase + r;
            vrow[r] = (row < ne);
            int rc = vrow[r] ? row : (ne - 1);
            toks[r] = list[rc];
        }
#pragma unroll
        for (int ns = 0; ns < 4; ns++) {
            int ncol = nb * 256 + wn * 64 + ns * 16 + l16;
            float bias = eb[(size_t)e * OUTD + ncol];
#pragma unroll
            for (int r = 0; r < 8; r++) {
                float v = acc[ms][ns][r] + bias;
                v = (v > 0.f) ? v * 0.5f : 0.f;
                if (vrow[r]) atomicAdd(out + (size_t)toks[r] * OUTD + ncol, v);
            }
        }
    }
}

extern "C" void kernel_launch(void* const* d_in, const int* in_sizes, int n_in,
                              void* d_out, int out_size, void* d_ws, size_t ws_size,
                              hipStream_t stream) {
    const float* x        = (const float*)d_in[0];
    const float* gate_w   = (const float*)d_in[1];
    const float* gate_b   = (const float*)d_in[2];
    const float* expert_w = (const float*)d_in[3];
    const float* expert_b = (const float*)d_in[4];
    float* out = (float*)d_out;
    char* ws = (char*)d_ws;
    int*       counts = (int*)(ws + WS_COUNTS);
    int*       lists  = (int*)(ws + WS_LISTS);
    _Float16*  xf     = (_Float16*)(ws + WS_XF16);
    _Float16*  wpk    = (_Float16*)(ws + WS_WPACK);

    (void)in_sizes; (void)n_in; (void)out_size; (void)ws_size;

    // zero output + counters
    k_init<<<(NTOK * (size_t)OUTD / 4 + 255) / 256, 256, 0, stream>>>(out, counts);
    // convert x to f16
    k_cvt_x<<<((size_t)NTOK * DIM / 4 + 255) / 256, 256, 0, stream>>>(x, xf);
    // pack expert weights into WMMA B fragments (f16)
    k_pack_w<<<((size_t)NEXP * DIM * OUTD + 255) / 256, 256, 0, stream>>>(expert_w, wpk);
    // gate + top-2 routing
    k_gate<<<NTOK / 8, 256, 0, stream>>>(x, gate_w, gate_b, counts, lists);
    // sparse expert GEMM with WMMA (64x64 wave tile, 16 WMMA : 16 loads per K-step)
    k_moe_gemm<<<NEXP * 128 * 8, 256, 0, stream>>>(xf, wpk, expert_b, counts, lists, out);
}
